// HybridNeuralBlock_27685359190826
// MI455X (gfx1250) — compile-verified, matrix operand order checked by reading, not run on hardware
//
#include <hip/hip_runtime.h>
#include <math.h>

typedef __attribute__((ext_vector_type(16))) _Float16 v16h;
typedef __attribute__((ext_vector_type(8)))  float    v8f;
typedef __attribute__((ext_vector_type(4)))  unsigned int u32x4;   // 16B vector
typedef _Float16 h16;

namespace {
constexpr int Bb = 2, Ss = 2048, Dm = 768, Hh = 12, HDs = 64;
constexpr int DI = 1536, NSTc = 16, DTRc = 48, KCc = 4, CAPc = 256, DFc = 3072;
constexpr int NTOK = Bb * Ss;  // 4096
}

// ---------------------------------------------------------------------------
// Activation epilogue
// ---------------------------------------------------------------------------
enum { ACT_NONE = 0, ACT_SILU = 1, ACT_GELU = 2, ACT_SIGMOID = 3, ACT_RELU = 4, ACT_SOFTPLUS = 5 };

__device__ __forceinline__ float apply_act(float v, int act) {
  switch (act) {
    case ACT_SILU:     return v / (1.0f + __expf(-v));
    case ACT_GELU:     return 0.5f * v * (1.0f + erff(v * 0.70710678118654752f));
    case ACT_SIGMOID:  return 1.0f / (1.0f + __expf(-v));
    case ACT_RELU:     return v > 0.0f ? v : 0.0f;
    case ACT_SOFTPLUS: return v > 20.0f ? v : log1pf(__expf(v));
    default:           return v;
  }
}

// ---------------------------------------------------------------------------
// Generic WMMA GEMM: C[M,N] = act(A[M,K] * Bt[N,K]^T + bias)
//  - A  row-major [M][K] f16
//  - Bt row-major [N][K] f16 (weights pre-transposed once at conversion time,
//    so the B tile load is identical to the A tile load: contiguous b128s)
// Block = 128 threads (4 waves), block tile 128x64.
// Wave w owns rows w*32..w*32+31 -> 2 A-frags, 4 B-frags, 8 WMMAs per k-step.
// CDNA5 f16 A-frag lane mapping (ISA 7.12.2): lane l: row m = l&15, hi = l>>4,
//   element e (0..15) holds k = e + 8*((e>>3) + hi).  B-frag mirrors A with
//   "row" = output column. C-frag: VGPR r holds row r + 8*hi, col = l&15.
// ---------------------------------------------------------------------------
__global__ void __launch_bounds__(128) gemm_wmma_k(
    const h16* __restrict__ A, int lda,
    const h16* __restrict__ Bt,            // [N][K], stride K
    float* __restrict__ Cf, int ldcf,
    h16* __restrict__ Ch, int ldch,
    const float* __restrict__ bias,
    int M, int N, int K, int act)
{
  __shared__ h16 As[128][40];  // 128 rows x 32 k (pad to 40 halves = 80B, 16B-aligned rows)
  __shared__ h16 Bs[64][40];   //  64 cols x 32 k
  const int tid  = threadIdx.x;
  const int w    = tid >> 5;
  const int lane = tid & 31;
  const int lm   = lane & 15;
  const int hi   = lane >> 4;
  const int n0   = blockIdx.x * 64;
  const int m0   = blockIdx.y * 128;
  const bool fullN = (n0 + 64 <= N);

  v8f acc[2][4] = {{v8f{}, v8f{}, v8f{}, v8f{}}, {v8f{}, v8f{}, v8f{}, v8f{}}};

  for (int k0 = 0; k0 < K; k0 += 32) {
    const bool fullK = (k0 + 32 <= K);
    __syncthreads();
    // ---- A tile (128x32) ----
    if (fullK && (m0 + 128 <= M)) {
      // 128*32/8 = 512 b128 slots, 4 per thread
      for (int i = tid; i < 512; i += 128) {
        int r = i >> 2, c = (i & 3) * 8;
        *(u32x4*)&As[r][c] = *(const u32x4*)&A[(long)(m0 + r) * lda + k0 + c];
      }
      if (k0 + 32 < K) __builtin_prefetch(&A[(long)(m0 + tid) * lda + k0 + 32], 0, 1);
    } else {
      for (int i = tid; i < 128 * 32; i += 128) {
        int r = i >> 5, c = i & 31;
        int gm = m0 + r, gk = k0 + c;
        As[r][c] = (gm < M && gk < K) ? A[(long)gm * lda + gk] : (h16)0.0f;
      }
    }
    // ---- B tile (64 cols x 32 k), same layout as A thanks to pre-transpose ----
    if (fullK && fullN) {
      for (int i = tid; i < 256; i += 128) {
        int r = i >> 2, c = (i & 3) * 8;
        *(u32x4*)&Bs[r][c] = *(const u32x4*)&Bt[(long)(n0 + r) * K + k0 + c];
      }
      if (k0 + 32 < K) __builtin_prefetch(&Bt[(long)(n0 + (tid & 63)) * K + k0 + 32], 0, 1);
    } else {
      for (int i = tid; i < 64 * 32; i += 128) {
        int r = i >> 5, c = i & 31;
        int gn = n0 + r, gk = k0 + c;
        Bs[r][c] = (gn < N && gk < K) ? Bt[(long)gn * K + gk] : (h16)0.0f;
      }
    }
    __syncthreads();

    v16h a0, a1;
    {
      const h16* ar0 = &As[w * 32 + lm][0];
      const h16* ar1 = &As[w * 32 + 16 + lm][0];
#pragma unroll
      for (int e = 0; e < 16; ++e) {
        int k = e + 8 * (e >> 3) + hi * 8;
        a0[e] = ar0[k];
        a1[e] = ar1[k];
      }
    }
#pragma unroll
    for (int nt = 0; nt < 4; ++nt) {
      v16h bb;
      const h16* brow = &Bs[nt * 16 + lm][0];
#pragma unroll
      for (int e = 0; e < 16; ++e) bb[e] = brow[e + 8 * (e >> 3) + hi * 8];
      acc[0][nt] = __builtin_amdgcn_wmma_f32_16x16x32_f16(
          false, a0, false, bb, (short)0, acc[0][nt], false, false);
      acc[1][nt] = __builtin_amdgcn_wmma_f32_16x16x32_f16(
          false, a1, false, bb, (short)0, acc[1][nt], false, false);
    }
  }

  // epilogue
#pragma unroll
  for (int mt = 0; mt < 2; ++mt) {
#pragma unroll
    for (int nt = 0; nt < 4; ++nt) {
      int gn = n0 + nt * 16 + lm;
      if (gn >= N) continue;
      float bv = bias ? bias[gn] : 0.0f;
#pragma unroll
      for (int r = 0; r < 8; ++r) {
        int gm = m0 + w * 32 + mt * 16 + r + hi * 8;
        if (gm >= M) continue;
        float v = apply_act(acc[mt][nt][r] + bv, act);
        if (Cf) Cf[(long)gm * ldcf + gn] = v;
        if (Ch) Ch[(long)gm * ldch + gn] = (h16)v;
      }
    }
  }
}

// ---------------------------------------------------------------------------
// Flash attention (no mask). One block = 64 query rows of one (b,h).
// 128 threads = 4 waves, wave w owns q rows w*16..w*16+15 (full HD=64).
// Streams over 32 key blocks of 64; online softmax; P.V via LDS refragment.
// ---------------------------------------------------------------------------
__global__ void __launch_bounds__(128) flash_attn_k(
    const h16* __restrict__ qh, const h16* __restrict__ kh,
    const h16* __restrict__ vh, h16* __restrict__ aoh)
{
  __shared__ h16 Ks[64][72];   // [key][d]
  __shared__ h16 Vt[64][72];   // [d][key]
  __shared__ h16 Ps[64][72];   // [q row][key]
  const int tid  = threadIdx.x;
  const int w    = tid >> 5;
  const int lane = tid & 31;
  const int lm   = lane & 15;
  const int hi   = lane >> 4;
  const int qb   = blockIdx.x;            // 0..31
  const int head = blockIdx.y;            // b*H + h
  const int b    = head / Hh;
  const int h    = head % Hh;
  const long hb  = (long)head * Ss * HDs;
  const float scale = 0.125f;             // HD^-0.5

  // Q fragments (held across the whole loop)
  v16h a0, a1;
  {
    const h16* qrow = qh + hb + (long)(qb * 64 + w * 16 + lm) * HDs;
#pragma unroll
    for (int e = 0; e < 16; ++e) {
      int k = e + 8 * (e >> 3) + hi * 8;
      a0[e] = qrow[k];
      a1[e] = qrow[k + 32];
    }
  }

  float rmax[8], rsum[8];
  v8f o[4] = {v8f{}, v8f{}, v8f{}, v8f{}};
#pragma unroll
  for (int r = 0; r < 8; ++r) { rmax[r] = -1e30f; rsum[r] = 0.0f; }

  for (int kb = 0; kb < 32; ++kb) {
    __syncthreads();
    // vectorized K/V block load: 64x64 halves = 512 b128 slots, 4 per thread
    for (int i = tid; i < 512; i += 128) {
      int key = i >> 3;
      int d0  = (i & 7) * 8;
      long g = hb + (long)(kb * 64 + key) * HDs + d0;
      u32x4 kv = *(const u32x4*)(kh + g);
      *(u32x4*)&Ks[key][d0] = kv;
      u32x4 vv = *(const u32x4*)(vh + g);
      const h16* vp = (const h16*)&vv;
#pragma unroll
      for (int j = 0; j < 8; ++j) Vt[d0 + j][key] = vp[j];
    }
    __syncthreads();

    // S = Q K^T
    v8f s[4];
#pragma unroll
    for (int nt = 0; nt < 4; ++nt) {
      v16h b0, b1;
      const h16* krow = &Ks[nt * 16 + lm][0];
#pragma unroll
      for (int e = 0; e < 16; ++e) {
        int k = e + 8 * (e >> 3) + hi * 8;
        b0[e] = krow[k];
        b1[e] = krow[k + 32];
      }
      v8f sa = {};
      sa = __builtin_amdgcn_wmma_f32_16x16x32_f16(false, a0, false, b0, (short)0, sa, false, false);
      sa = __builtin_amdgcn_wmma_f32_16x16x32_f16(false, a1, false, b1, (short)0, sa, false, false);
      s[nt] = sa;
    }

    // online softmax, per accumulator row r (row = w*16 + r + 8*hi)
#pragma unroll
    for (int r = 0; r < 8; ++r) {
      float m = fmaxf(fmaxf(s[0][r], s[1][r]), fmaxf(s[2][r], s[3][r])) * scale;
#pragma unroll
      for (int mk = 1; mk <= 8; mk <<= 1) m = fmaxf(m, __shfl_xor(m, mk, 32));
      float nm = fmaxf(rmax[r], m);
      float alpha = __expf(rmax[r] - nm);
      rmax[r] = nm;
      float rs = 0.0f;
#pragma unroll
      for (int nt = 0; nt < 4; ++nt) {
        float pv = __expf(s[nt][r] * scale - nm);
        s[nt][r] = pv;
        rs += pv;
      }
#pragma unroll
      for (int mk = 1; mk <= 8; mk <<= 1) rs += __shfl_xor(rs, mk, 32);
      rsum[r] = rsum[r] * alpha + rs;
#pragma unroll
      for (int dt = 0; dt < 4; ++dt) o[dt][r] *= alpha;
    }

    // stage P (f16) in this wave's private LDS rows
#pragma unroll
    for (int nt = 0; nt < 4; ++nt)
#pragma unroll
      for (int r = 0; r < 8; ++r)
        Ps[w * 16 + r + hi * 8][nt * 16 + lm] = (h16)s[nt][r];

    // O += P V  (same-wave LDS RAW, in-order DS pipe)
    v16h p0, p1;
    {
      const h16* prow = &Ps[w * 16 + lm][0];
#pragma unroll
      for (int e = 0; e < 16; ++e) {
        int k = e + 8 * (e >> 3) + hi * 8;
        p0[e] = prow[k];
        p1[e] = prow[k + 32];
      }
    }
#pragma unroll
    for (int dt = 0; dt < 4; ++dt) {
      v16h vb0, vb1;
      const h16* vrow = &Vt[dt * 16 + lm][0];
#pragma unroll
      for (int e = 0; e < 16; ++e) {
        int k = e + 8 * (e >> 3) + hi * 8;
        vb0[e] = vrow[k];
        vb1[e] = vrow[k + 32];
      }
      o[dt] = __builtin_amdgcn_wmma_f32_16x16x32_f16(false, p0, false, vb0, (short)0, o[dt], false, false);
      o[dt] = __builtin_amdgcn_wmma_f32_16x16x32_f16(false, p1, false, vb1, (short)0, o[dt], false, false);
    }
  }

  // normalize + write (b, s, h*64+d) as f16 for the output projection GEMM
#pragma unroll
  for (int dt = 0; dt < 4; ++dt)
#pragma unroll
    for (int r = 0; r < 8; ++r) {
      int sidx = qb * 64 + w * 16 + r + hi * 8;
      float val = o[dt][r] / rsum[r];
      aoh[((long)(b * Ss + sidx)) * Dm + h * HDs + dt * 16 + lm] = (h16)val;
    }
}

// ---------------------------------------------------------------------------
// Elementwise / small kernels
// ---------------------------------------------------------------------------
// Convert f32 weight [K][N] -> f16 transposed [N][K] (coalesced reads)
__global__ void cvt_transpose_k(const float* __restrict__ s, h16* __restrict__ d, int K, int N) {
  int t = blockIdx.x * 256 + threadIdx.x;
  if (t >= K * N) return;
  int n = t % N, k = t / N;
  d[(long)n * K + k] = (h16)s[t];
}

__global__ void __launch_bounds__(256) layernorm_k(
    const float* __restrict__ x, const float* __restrict__ w, const float* __restrict__ b,
    float* __restrict__ yf, h16* __restrict__ yh)
{
  __shared__ float s1[256], s2[256];
  const int tid = threadIdx.x;
  const long row = blockIdx.x;
  const float* xr = x + row * Dm;
  float v0 = xr[tid], v1 = xr[tid + 256], v2 = xr[tid + 512];
  s1[tid] = v0 + v1 + v2;
  s2[tid] = v0 * v0 + v1 * v1 + v2 * v2;
  __syncthreads();
  for (int st = 128; st > 0; st >>= 1) {
    if (tid < st) { s1[tid] += s1[tid + st]; s2[tid] += s2[tid + st]; }
    __syncthreads();
  }
  const float mu = s1[0] * (1.0f / Dm);
  const float var = s2[0] * (1.0f / Dm) - mu * mu;
  const float rs = rsqrtf(var + 1e-5f);
  float vs[3] = {v0, v1, v2};
#pragma unroll
  for (int j = 0; j < 3; ++j) {
    int c = tid + j * 256;
    float y = (vs[j] - mu) * rs * w[c] + b[c];
    if (yf) yf[row * Dm + c] = y;
    if (yh) yh[row * Dm + c] = (h16)y;
  }
}

__global__ void rope_split_k(const float* __restrict__ qkv, h16* __restrict__ qh,
                             h16* __restrict__ kh, h16* __restrict__ vh)
{
  int t = blockIdx.x * 256 + threadIdx.x;
  if (t >= NTOK * Hh * 32) return;
  int i = t & 31;
  int hh = (t >> 5) % Hh;
  int tok = t / (32 * Hh);
  int b = tok / Ss, s = tok % Ss;
  float inv = __powf(10000.0f, -(float)(2 * i) * (1.0f / HDs));
  float f = (float)s * inv;
  float c = cosf(f), sn = sinf(f);
  const float* base = qkv + (long)tok * (3 * Dm) + hh * HDs;
  float q1 = base[i], q2 = base[i + 32];
  float k1 = base[Dm + i], k2 = base[Dm + i + 32];
  long o = ((long)(b * Hh + hh) * Ss + s) * HDs;
  qh[o + i]      = (h16)(q1 * c - q2 * sn);
  qh[o + 32 + i] = (h16)(q1 * sn + q2 * c);
  kh[o + i]      = (h16)(k1 * c - k2 * sn);
  kh[o + 32 + i] = (h16)(k1 * sn + k2 * c);
  vh[o + i]      = (h16)base[2 * Dm + i];
  vh[o + 32 + i] = (h16)base[2 * Dm + i + 32];
}

__global__ void conv_silu_k(const float* __restrict__ xz, const float* __restrict__ cw,
                            const float* __restrict__ cb, float* __restrict__ uf,
                            h16* __restrict__ uh, float* __restrict__ zsf)
{
  int t = blockIdx.x * 256 + threadIdx.x;
  if (t >= NTOK * DI) return;
  int d = t % DI, tok = t / DI;
  int b = tok / Ss, s = tok % Ss;
  float acc = cb[d];
#pragma unroll
  for (int j = 0; j < KCc; ++j) {
    int ss = s + j - (KCc - 1);
    if (ss >= 0) acc += xz[((long)b * Ss + ss) * (2 * DI) + d] * cw[d * KCc + j];
  }
  float uu = acc / (1.0f + __expf(-acc));
  uf[(long)tok * DI + d] = uu;
  uh[(long)tok * DI + d] = (h16)uu;
  float zz = xz[(long)tok * (2 * DI) + DI + d];
  zsf[(long)tok * DI + d] = zz / (1.0f + __expf(-zz));
}

__global__ void dlow_cvt_k(const float* __restrict__ dbc, h16* __restrict__ dlow) {
  int t = blockIdx.x * 256 + threadIdx.x;
  if (t >= NTOK * DTRc) return;
  int c = t % DTRc, tok = t / DTRc;
  dlow[t] = (h16)dbc[(long)tok * 80 + c];
}

__global__ void __launch_bounds__(256) ssm_scan_k(
    const float* __restrict__ delta, const float* __restrict__ u,
    const float* __restrict__ dbc, const float* __restrict__ zs,
    const float* __restrict__ A_log, const float* __restrict__ Dp,
    h16* __restrict__ ys)
{
  const int d = blockIdx.x * 256 + threadIdx.x;  // channel 0..DI-1
  const int b = blockIdx.y;
  float Ar[NSTc];
#pragma unroll
  for (int n = 0; n < NSTc; ++n) Ar[n] = -__expf(A_log[d * NSTc + n]);
  const float Dd = Dp[d];
  float h[NSTc];
#pragma unroll
  for (int n = 0; n < NSTc; ++n) h[n] = 0.0f;
  for (int s = 0; s < Ss; ++s) {
    const long tok = (long)b * Ss + s;
    const float dt = delta[tok * DI + d];
    const float uu = u[tok * DI + d];
    const float du = dt * uu;
    const float* bc = dbc + tok * 80;
    float y = 0.0f;
#pragma unroll
    for (int n = 0; n < NSTc; ++n) {
      h[n] = __expf(dt * Ar[n]) * h[n] + du * bc[DTRc + n];
      y += h[n] * bc[DTRc + NSTc + n];
    }
    y += uu * Dd;
    ys[tok * DI + d] = (h16)(y * zs[tok * DI + d]);
  }
}

__global__ void __launch_bounds__(256) combine_ln2_k(
    const float* __restrict__ x, const float* __restrict__ g,
    const float* __restrict__ at, const float* __restrict__ sm,
    const float* __restrict__ w, const float* __restrict__ b,
    float* __restrict__ x1f, float* __restrict__ xn2f, h16* __restrict__ xn2h,
    float* __restrict__ outp)
{
  __shared__ float s1[256], s2[256];
  const int tid = threadIdx.x;
  const long row = blockIdx.x;
  float xv[3];
  float ssum = 0.0f, ssq = 0.0f;
#pragma unroll
  for (int j = 0; j < 3; ++j) {
    int c = tid + j * 256;
    long o = row * Dm + c;
    float gv = g[o];
    float v = x[o] + gv * at[o] + (1.0f - gv) * sm[o];
    x1f[o] = v;
    xv[j] = v;
    ssum += v; ssq += v * v;
  }
  s1[tid] = ssum; s2[tid] = ssq;
  __syncthreads();
  for (int st = 128; st > 0; st >>= 1) {
    if (tid < st) { s1[tid] += s1[tid + st]; s2[tid] += s2[tid + st]; }
    __syncthreads();
  }
  float mu = s1[0] * (1.0f / Dm);
  float var = s2[0] * (1.0f / Dm) - mu * mu;
  float rs = rsqrtf(var + 1e-5f);
#pragma unroll
  for (int j = 0; j < 3; ++j) {
    int c = tid + j * 256;
    long o = row * Dm + c;
    float y = (xv[j] - mu) * rs * w[c] + b[c];
    xn2f[o] = y;
    xn2h[o] = (h16)y;
    outp[o] = xv[j] + y;   // baseline out = x1 + xn2; selected rows overwritten later
  }
}

__global__ void __launch_bounds__(64) router_score_k(
    const float* __restrict__ hr, const float* __restrict__ w2,
    const float* __restrict__ b2, float* __restrict__ sc)
{
  __shared__ float red[64];
  const int tid = threadIdx.x;
  const long row = blockIdx.x;
  float s = 0.0f;
  for (int c = tid; c < Dm / 4; c += 64) s += hr[row * (Dm / 4) + c] * w2[c];
  red[tid] = s;
  __syncthreads();
  for (int st = 32; st > 0; st >>= 1) {
    if (tid < st) red[tid] += red[tid + st];
    __syncthreads();
  }
  if (tid == 0) sc[row] = red[0] + b2[0];
}

// iterative top-CAP selection, stable lower-index tie-break (matches lax.top_k)
__global__ void __launch_bounds__(256) topk_k(const float* __restrict__ sc, int* __restrict__ idx)
{
  __shared__ float vals[Ss];
  __shared__ float bv[256];
  __shared__ int   bi[256];
  const int tid = threadIdx.x;
  const int b = blockIdx.x;
  for (int i = tid; i < Ss; i += 256) vals[i] = sc[(long)b * Ss + i];
  __syncthreads();
  for (int it = 0; it < CAPc; ++it) {
    float best = -3.0e38f; int bidx = 1 << 30;
    for (int i = tid; i < Ss; i += 256) {
      float v = vals[i];
      if (v > best || (v == best && i < bidx)) { best = v; bidx = i; }
    }
    bv[tid] = best; bi[tid] = bidx;
    __syncthreads();
    for (int st = 128; st > 0; st >>= 1) {
      if (tid < st) {
        float v = bv[tid + st]; int ii = bi[tid + st];
        if (v > bv[tid] || (v == bv[tid] && ii < bi[tid])) { bv[tid] = v; bi[tid] = ii; }
      }
      __syncthreads();
    }
    if (tid == 0) { idx[b * CAPc + it] = bi[0]; vals[bi[0]] = -3.4e38f; }
    __syncthreads();
  }
}

__global__ void gather_sel_k(const float* __restrict__ xn2, const int* __restrict__ idx,
                             h16* __restrict__ sel)
{
  int t = blockIdx.x * 256 + threadIdx.x;
  if (t >= Bb * CAPc * Dm) return;
  int c = t % Dm, r = t / Dm;
  int b = r / CAPc;
  sel[(long)r * Dm + c] = (h16)xn2[((long)b * Ss + idx[r]) * Dm + c];
}

__global__ void scatter_ffn_k(const float* __restrict__ x1, const float* __restrict__ proc,
                              const int* __restrict__ idx, float* __restrict__ out)
{
  int t = blockIdx.x * 256 + threadIdx.x;
  if (t >= Bb * CAPc * Dm) return;
  int c = t % Dm, r = t / Dm;
  int b = r / CAPc;
  long row = (long)b * Ss + idx[r];
  out[row * Dm + c] = x1[row * Dm + c] + proc[(long)r * Dm + c];
}

// ---------------------------------------------------------------------------
// Host driver
// ---------------------------------------------------------------------------
extern "C" void kernel_launch(void* const* d_in, const int* in_sizes, int n_in,
                              void* d_out, int out_size, void* d_ws, size_t ws_size,
                              hipStream_t stream) {
  (void)in_sizes; (void)n_in; (void)out_size; (void)ws_size;
  const float* x          = (const float*)d_in[0];
  const float* ln1_w      = (const float*)d_in[1];
  const float* ln1_b      = (const float*)d_in[2];
  const float* ln2_w      = (const float*)d_in[3];
  const float* ln2_b      = (const float*)d_in[4];
  const float* qkv_w      = (const float*)d_in[5];
  const float* attn_out_w = (const float*)d_in[6];
  const float* attn_out_b = (const float*)d_in[7];
  const float* in_proj_w  = (const float*)d_in[8];
  const float* conv_w     = (const float*)d_in[9];
  const float* conv_b     = (const float*)d_in[10];
  const float* x_proj_w   = (const float*)d_in[11];
  const float* dt_w       = (const float*)d_in[12];
  const float* dt_b       = (const float*)d_in[13];
  const float* A_log      = (const float*)d_in[14];
  const float* Dvec       = (const float*)d_in[15];
  const float* ssm_out_w  = (const float*)d_in[16];
  const float* gate_w     = (const float*)d_in[17];
  const float* gate_b     = (const float*)d_in[18];
  const float* ffn_w1     = (const float*)d_in[19];
  const float* ffn_b1     = (const float*)d_in[20];
  const float* ffn_w2     = (const float*)d_in[21];
  const float* ffn_b2     = (const float*)d_in[22];
  const float* r_w1       = (const float*)d_in[23];
  const float* r_b1       = (const float*)d_in[24];
  const float* r_w2       = (const float*)d_in[25];
  const float* r_b2       = (const float*)d_in[26];
  float* out = (float*)d_out;

  // workspace carve-out
  char* p = (char*)d_ws;
  auto alloc = [&](size_t bytes) { char* r = p; p += (bytes + 255) & ~(size_t)255; return r; };

  h16* xn_h     = (h16*)alloc((size_t)NTOK * Dm * 2);
  h16* qkvw_h   = (h16*)alloc((size_t)Dm * 3 * Dm * 2);      // [2304][768]
  h16* attnw_h  = (h16*)alloc((size_t)Dm * Dm * 2);          // [768][768]
  h16* inprjw_h = (h16*)alloc((size_t)Dm * 2 * DI * 2);      // [3072][768]
  h16* xprjw_h  = (h16*)alloc((size_t)DI * 80 * 2);          // [80][1536]
  h16* dtw_h    = (h16*)alloc((size_t)DTRc * DI * 2);        // [1536][48]
  h16* ssmw_h   = (h16*)alloc((size_t)DI * Dm * 2);          // [768][1536]
  h16* gatew_h  = (h16*)alloc((size_t)2 * Dm * Dm * 2);      // [768][1536]
  h16* ffn1w_h  = (h16*)alloc((size_t)Dm * DFc * 2);         // [3072][768]
  h16* ffn2w_h  = (h16*)alloc((size_t)DFc * Dm * 2);         // [768][3072]
  h16* r1w_h    = (h16*)alloc((size_t)Dm * (Dm / 4) * 2);    // [192][768]
  h16* q_h      = (h16*)alloc((size_t)Bb * Hh * Ss * HDs * 2);
  h16* k_h      = (h16*)alloc((size_t)Bb * Hh * Ss * HDs * 2);
  h16* v_h      = (h16*)alloc((size_t)Bb * Hh * Ss * HDs * 2);
  h16* ao_h     = (h16*)alloc((size_t)NTOK * Dm * 2);
  h16* cat_h    = (h16*)alloc((size_t)NTOK * 2 * Dm * 2);
  h16* u_h      = (h16*)alloc((size_t)NTOK * DI * 2);
  h16* dlow_h   = (h16*)alloc((size_t)NTOK * DTRc * 2);
  h16* ys_h     = (h16*)alloc((size_t)NTOK * DI * 2);
  h16* xn2_h    = (h16*)alloc((size_t)NTOK * Dm * 2);
  h16* sel_h    = (h16*)alloc((size_t)Bb * CAPc * Dm * 2);
  h16* hid_h    = (h16*)alloc((size_t)Bb * CAPc * DFc * 2);

  float* xz_f   = (float*)alloc((size_t)NTOK * 2 * DI * 4);  // also aliases qkv_f (dead after rope)
  float* qkv_f  = xz_f;                                      // NTOK*2304 <= NTOK*3072
  float* attn_f = (float*)alloc((size_t)NTOK * Dm * 4);
  float* dbc_f  = (float*)alloc((size_t)NTOK * 80 * 4);
  float* delta_f= (float*)alloc((size_t)NTOK * DI * 4);
  float* u_f    = (float*)alloc((size_t)NTOK * DI * 4);
  float* zs_f   = (float*)alloc((size_t)NTOK * DI * 4);
  float* ssm_f  = (float*)alloc((size_t)NTOK * Dm * 4);
  float* g_f    = (float*)alloc((size_t)NTOK * Dm * 4);
  float* x1_f   = (float*)alloc((size_t)NTOK * Dm * 4);
  float* xn2_f  = (float*)alloc((size_t)NTOK * Dm * 4);
  float* hr_f   = (float*)alloc((size_t)NTOK * (Dm / 4) * 4);
  float* sc_f   = (float*)alloc((size_t)NTOK * 4);
  float* proc_f = (float*)alloc((size_t)Bb * CAPc * Dm * 4);
  int*   idx_i  = (int*)alloc((size_t)Bb * CAPc * 4);

  auto cvtT = [&](const float* s, h16* d, int Kdim, int Ndim) {
    int n = Kdim * Ndim;
    cvt_transpose_k<<<(n + 255) / 256, 256, 0, stream>>>(s, d, Kdim, Ndim);
  };
  auto gemm = [&](const h16* A, int lda, const h16* Bt,
                  float* Cf, int ldcf, h16* Ch, int ldch,
                  const float* bias, int M, int N, int K, int act) {
    dim3 grid((N + 63) / 64, (M + 127) / 128);
    gemm_wmma_k<<<grid, 128, 0, stream>>>(A, lda, Bt, Cf, ldcf, Ch, ldch, bias, M, N, K, act);
  };

  // weights -> f16, transposed to [N][K] (in-stream, graph-capture safe)
  cvtT(qkv_w, qkvw_h, Dm, 3 * Dm);
  cvtT(attn_out_w, attnw_h, Dm, Dm);
  cvtT(in_proj_w, inprjw_h, Dm, 2 * DI);
  cvtT(x_proj_w, xprjw_h, DI, 80);
  cvtT(dt_w, dtw_h, DTRc, DI);
  cvtT(ssm_out_w, ssmw_h, DI, Dm);
  cvtT(gate_w, gatew_h, 2 * Dm, Dm);
  cvtT(ffn_w1, ffn1w_h, Dm, DFc);
  cvtT(ffn_w2, ffn2w_h, DFc, Dm);
  cvtT(r_w1, r1w_h, Dm, Dm / 4);

  // 1) LN1 -> xn (f16)
  layernorm_k<<<NTOK, 256, 0, stream>>>(x, ln1_w, ln1_b, nullptr, xn_h);
  // 2) qkv = xn @ qkv_w  (4096 x 2304 x 768)
  gemm(xn_h, Dm, qkvw_h, qkv_f, 3 * Dm, nullptr, 0, nullptr, NTOK, 3 * Dm, Dm, ACT_NONE);
  // 3) RoPE + head split -> q/k/v f16 (b,h,s,d)
  rope_split_k<<<(NTOK * Hh * 32 + 255) / 256, 256, 0, stream>>>(qkv_f, q_h, k_h, v_h);
  // 4) flash attention -> ao (f16, token-major)
  flash_attn_k<<<dim3(Ss / 64, Bb * Hh), 128, 0, stream>>>(q_h, k_h, v_h, ao_h);
  // 5) attn_out = ao @ attn_out_w + b (f32) and f16 into cat[:, :768]
  gemm(ao_h, Dm, attnw_h, attn_f, Dm, cat_h, 2 * Dm, attn_out_b, NTOK, Dm, Dm, ACT_NONE);
  // 6) xz = xn @ in_proj_w  (4096 x 3072 x 768)  [overwrites qkv_f alias]
  gemm(xn_h, Dm, inprjw_h, xz_f, 2 * DI, nullptr, 0, nullptr, NTOK, 2 * DI, Dm, ACT_NONE);
  // 7) causal conv + silu -> u (f32+f16), silu(z) -> zs
  conv_silu_k<<<(NTOK * DI + 255) / 256, 256, 0, stream>>>(xz_f, conv_w, conv_b, u_f, u_h, zs_f);
  // 8) dbc = u @ x_proj_w  (4096 x 80 x 1536)
  gemm(u_h, DI, xprjw_h, dbc_f, 80, nullptr, 0, nullptr, NTOK, 80, DI, ACT_NONE);
  // 9) dlow f16
  dlow_cvt_k<<<(NTOK * DTRc + 255) / 256, 256, 0, stream>>>(dbc_f, dlow_h);
  // 10) delta = softplus(dlow @ dt_w + dt_b)  (4096 x 1536 x 48)
  gemm(dlow_h, DTRc, dtw_h, delta_f, DI, nullptr, 0, dt_b, NTOK, DI, DTRc, ACT_SOFTPLUS);
  // 11) selective-scan (serial over S, parallel over B*DI channels)
  ssm_scan_k<<<dim3(DI / 256, Bb), 256, 0, stream>>>(delta_f, u_f, dbc_f, zs_f, A_log, Dvec, ys_h);
  // 12) ssm_out = ys @ ssm_out_w (f32) and f16 into cat[:, 768:]
  gemm(ys_h, DI, ssmw_h, ssm_f, Dm, cat_h + Dm, 2 * Dm, nullptr, NTOK, Dm, DI, ACT_NONE);
  // 13) g = sigmoid(cat @ gate_w + gate_b)
  gemm(cat_h, 2 * Dm, gatew_h, g_f, Dm, nullptr, 0, gate_b, NTOK, Dm, 2 * Dm, ACT_SIGMOID);
  // 14) x1 = x + g*attn + (1-g)*ssm ; LN2 ; baseline out = x1 + xn2
  combine_ln2_k<<<NTOK, 256, 0, stream>>>(x, g_f, attn_f, ssm_f, ln2_w, ln2_b,
                                          x1_f, xn2_f, xn2_h, out);
  // 15) router hidden = relu(xn2 @ r_w1 + r_b1)
  gemm(xn2_h, Dm, r1w_h, hr_f, Dm / 4, nullptr, 0, r_b1, NTOK, Dm / 4, Dm, ACT_RELU);
  // 16) score = hr @ r_w2 + r_b2
  router_score_k<<<NTOK, 64, 0, stream>>>(hr_f, r_w2, r_b2, sc_f);
  // 17) top-CAP per batch
  topk_k<<<Bb, 256, 0, stream>>>(sc_f, idx_i);
  // 18) gather selected rows of xn2 (f16)
  gather_sel_k<<<(Bb * CAPc * Dm + 255) / 256, 256, 0, stream>>>(xn2_f, idx_i, sel_h);
  // 19) hid = gelu(sel @ ffn_w1 + b1)   (512 x 3072 x 768)
  gemm(sel_h, Dm, ffn1w_h, nullptr, 0, hid_h, DFc, ffn_b1, Bb * CAPc, DFc, Dm, ACT_GELU);
  // 20) proc = hid @ ffn_w2 + b2        (512 x 768 x 3072)
  gemm(hid_h, DFc, ffn2w_h, proc_f, Dm, nullptr, 0, ffn_b2, Bb * CAPc, Dm, DFc, ACT_NONE);
  // 21) overwrite selected rows: out = x1 + proc
  scatter_ffn_k<<<(Bb * CAPc * Dm + 255) / 256, 256, 0, stream>>>(x1_f, proc_f, idx_i, out);
}
